// MixerBlock_38766374814177
// MI455X (gfx1250) — matrix operand order repaired and checked by reading, not compiled, not run on hardware
//
#include <hip/hip_runtime.h>
#include <hip/hip_bf16.h>
#include <math.h>

#define N_NODES 50000
#define NE_EDGE 400000
#define CDIM    128
#define HIDDIM  256
#define NTOK    64
#define NPAD    50176          // 196*256, padded row count for unguarded stores
#define CHUNK   256
#define NCHUNK  196            // ceil(50000/256)
#define GRID_E  3125           // 400000/128
#define GRID_ND 391            // ceil(50000/128)
#define GRID_N  782            // ceil(50000/64)

typedef _Float16 h16;
typedef __attribute__((ext_vector_type(16))) _Float16 v16h;
typedef __attribute__((ext_vector_type(8)))  _Float16 v8h;
typedef __attribute__((ext_vector_type(4)))  _Float16 v4h;
typedef __attribute__((ext_vector_type(8)))  float    v8f;

// ---------------- WMMA fragment helpers (CDNA5 16x16x32 f16) ----------------
// A (16xK, row-major): lane m=lane&15; halves 0..7 = A[m][k0+kOff+i],
// halves 8..15 = A[m][k0+kOff+16+i], kOff = (lane&16)?8:0.
__device__ __forceinline__ v16h frag_a(const h16* __restrict__ p, int stride,
                                       int m, int k0, int lane) {
  const int kOff = (lane & 16) ? 8 : 0;
  const h16* b0 = p + (size_t)m * stride + k0 + kOff;
  v8h lo = *(const v8h*)b0;
  v8h hi = *(const v8h*)(b0 + 16);
  return __builtin_shufflevector(lo, hi, 0,1,2,3,4,5,6,7,8,9,10,11,12,13,14,15);
}
// B (Kx16): lane holds column n; halves 0..7 = B[k0+kOff+i][n], 8..15 = B[k0+kOff+16+i][n], kOff=(lane&16)?8:0.
// For y = x @ W^T with W[n][k] row-major: column n of B == row n of W (contiguous).
__device__ __forceinline__ v16h frag_b(const h16* __restrict__ p, size_t stride,
                                       int row, int k0, int lane) {
  const int kOff = (lane & 16) ? 8 : 0;
  const h16* b0 = p + (size_t)row * stride + k0 + kOff;
  v8h lo = *(const v8h*)b0;
  v8h hi = *(const v8h*)(b0 + 16);
  return __builtin_shufflevector(lo, hi, 0,1,2,3,4,5,6,7,8,9,10,11,12,13,14,15);
}
__device__ __forceinline__ v8f wmma16(v16h a, v16h b, v8f c) {
  return __builtin_amdgcn_wmma_f32_16x16x32_f16(false, a, false, b, (short)0, c,
                                                false, false);
}
__device__ __forceinline__ float gelu_exact(float x) {
  return 0.5f * x * (1.0f + erff(x * 0.70710678118654752f));
}
__device__ __forceinline__ float silu(float x) { return x / (1.0f + __expf(-x)); }

// ---------------- utility kernels ----------------
__global__ void k_zero(float* __restrict__ p, long n) {
  long i = (long)blockIdx.x * blockDim.x + threadIdx.x;
  if (i < n) p[i] = 0.0f;
}
__global__ void k_cvt(const float* __restrict__ s, h16* __restrict__ d, int n) {
  int i = blockIdx.x * blockDim.x + threadIdx.x;
  if (i < n) d[i] = (h16)s[i];
}

// ---------------- edge MLP + LN + gate + scatter ----------------
// 128 edges / block, 256 threads (8 waves x 16-row strips).
// k-outer GEMMs: acc in VGPRs, A loaded once per k-step, weight k-slice staged
// in LDS once per block (read-once weight traffic).
__global__ __launch_bounds__(256)
void k_edge(const float* __restrict__ V, const float* __restrict__ E,
            const int* __restrict__ edges,
            const h16* __restrict__ w1, const float* __restrict__ b1,
            const h16* __restrict__ w2, const float* __restrict__ b2,
            const float* __restrict__ g, const float* __restrict__ be,
            const float* __restrict__ a2p,
            float* __restrict__ Eout, float* __restrict__ esum,
            float* __restrict__ cnt) {
  extern __shared__ char smem[];
  h16*   xh   = (h16*)smem;                         // [128][392] f16 (aliases outf)
  float* outf = (float*)smem;                       // [128][132] f32
  h16*   hh   = (h16*)(smem + 128 * 392 * 2);       // [128][264] f16
  h16*   wbuf = (h16*)(smem + 128 * 392 * 2 + 128 * 264 * 2); // [256][40] f16
  float* st   = (float*)(smem + 128 * 392 * 2 + 128 * 264 * 2 + 256 * 40 * 2);
  const int tid = threadIdx.x, lane = tid & 31, wave = tid >> 5;
  const int ebase = blockIdx.x * 128;
  __builtin_prefetch(w1, 0, 3);
  __builtin_prefetch(w2, 0, 3);

  // stage [sender | receiver | E] -> f16 LDS
  for (int i = tid; i < 128 * 96; i += 256) {
    int e = i / 96, c4 = (i % 96) * 4;
    const float* src;
    if (c4 < 128)      src = V + (size_t)edges[(size_t)(ebase + e) * 2 + 0] * CDIM + c4;
    else if (c4 < 256) src = V + (size_t)edges[(size_t)(ebase + e) * 2 + 1] * CDIM + (c4 - 128);
    else               src = E + (size_t)(ebase + e) * CDIM + (c4 - 256);
    float4 v = *(const float4*)src;
    v4h o; o[0] = (h16)v.x; o[1] = (h16)v.y; o[2] = (h16)v.z; o[3] = (h16)v.w;
    *(v4h*)(xh + (size_t)e * 392 + c4) = o;
  }

  const int mbase = wave * 16, nlane = lane & 15, roff = (lane & 16) ? 8 : 0;
  // GEMM1 [128x384]@[384->256] + GELU -> hh  (k-outer, acc in regs)
  {
    v8f acc[16];
#pragma unroll
    for (int nt = 0; nt < 16; ++nt) acc[nt] = (v8f){};
    for (int ks = 0; ks < 12; ++ks) {
      __syncthreads();                      // wbuf reuse + (ks==0) x-staging fence
      {                                     // stage w1[:, ks*32..+32) -> wbuf[256][40]
        const h16* src = w1 + (size_t)tid * 384 + ks * 32;
        h16* dst = wbuf + tid * 40;
        *(v8h*)(dst +  0) = *(const v8h*)(src +  0);
        *(v8h*)(dst +  8) = *(const v8h*)(src +  8);
        *(v8h*)(dst + 16) = *(const v8h*)(src + 16);
        *(v8h*)(dst + 24) = *(const v8h*)(src + 24);
      }
      __syncthreads();
      v16h a = frag_a(xh, 392, mbase + nlane, ks * 32, lane);
#pragma unroll
      for (int nt = 0; nt < 16; ++nt)
        acc[nt] = wmma16(a, frag_b(wbuf, 40, nt * 16 + nlane, 0, lane), acc[nt]);
    }
#pragma unroll
    for (int nt = 0; nt < 16; ++nt) {
      float bias = b1[nt * 16 + nlane];
      h16* dst = hh + (size_t)(mbase + roff) * 264 + nt * 16 + nlane;
#pragma unroll
      for (int r = 0; r < 8; ++r) dst[(size_t)r * 264] = (h16)gelu_exact(acc[nt][r] + bias);
    }
  }
  // GEMM2 [128x256]@[256->128] -> outf (aliases xh)
  {
    v8f acc[8];
#pragma unroll
    for (int nt = 0; nt < 8; ++nt) acc[nt] = (v8f){};
    for (int ks = 0; ks < 8; ++ks) {
      __syncthreads();
      {                                     // stage w2[:, ks*32..+32) (128 rows)
        int row = tid >> 1, part = (tid & 1) * 16;
        const h16* src = w2 + (size_t)row * 256 + ks * 32 + part;
        h16* dst = wbuf + row * 40 + part;
        *(v8h*)(dst + 0) = *(const v8h*)(src + 0);
        *(v8h*)(dst + 8) = *(const v8h*)(src + 8);
      }
      __syncthreads();
      v16h a = frag_a(hh, 264, mbase + nlane, ks * 32, lane);
#pragma unroll
      for (int nt = 0; nt < 8; ++nt)
        acc[nt] = wmma16(a, frag_b(wbuf, 40, nt * 16 + nlane, 0, lane), acc[nt]);
    }
#pragma unroll
    for (int nt = 0; nt < 8; ++nt) {
      float bias = b2[nt * 16 + nlane];
      float* dst = outf + (size_t)(mbase + roff) * 132 + nt * 16 + nlane;
#pragma unroll
      for (int r = 0; r < 8; ++r) dst[(size_t)r * 132] = acc[nt][r] + bias;
    }
  }
  __syncthreads();
  // LayerNorm stats
  for (int r = tid; r < 128; r += 256) {
    const float* row = outf + (size_t)r * 132;
    float s = 0.f, s2 = 0.f;
    for (int c = 0; c < 128; ++c) { float v = row[c]; s += v; s2 += v * v; }
    float mu = s * (1.0f / 128.0f);
    st[r] = mu;
    st[128 + r] = rsqrtf(s2 * (1.0f / 128.0f) - mu * mu + 1e-5f);
  }
  __syncthreads();
  const float sa2 = silu(a2p[0]);
  for (int i = tid; i < 128 * 32; i += 256) {
    int e = i / 32, c4 = (i % 32) * 4;
    int ge = ebase + e;
    int recv = edges[(size_t)ge * 2 + 1];
    float mu = st[e], rs = st[128 + e];
    float4 eo;
    float vv[4];
#pragma unroll
    for (int j = 0; j < 4; ++j) {
      int c = c4 + j;
      vv[j] = (outf[(size_t)e * 132 + c] - mu) * rs * g[c] + be[c];
      atomicAdd(&esum[(size_t)recv * CDIM + c], vv[j]);
    }
    const float* Er = E + (size_t)ge * CDIM + c4;
    eo.x = Er[0] + sa2 * vv[0]; eo.y = Er[1] + sa2 * vv[1];
    eo.z = Er[2] + sa2 * vv[2]; eo.w = Er[3] + sa2 * vv[3];
    *(float4*)(Eout + (size_t)ge * CDIM + c4) = eo;
  }
  if (tid < 128) atomicAdd(&cnt[edges[(size_t)(ebase + tid) * 2 + 1]], 1.0f);
}

// ---------------- node MLP + gated residual + LN1 ----------------
// 128 nodes / block, 256 threads (8 waves), same k-outer scheme.
__global__ __launch_bounds__(256)
void k_node(const float* __restrict__ V, const float* __restrict__ esum,
            const float* __restrict__ cnt,
            const h16* __restrict__ w1, const float* __restrict__ b1,
            const h16* __restrict__ w2, const float* __restrict__ b2,
            const float* __restrict__ g, const float* __restrict__ be,
            const float* __restrict__ ln1g, const float* __restrict__ ln1b,
            const float* __restrict__ a1p,
            float* __restrict__ Vr, h16* __restrict__ W0h) {
  extern __shared__ char smem[];
  h16*   xh   = (h16*)smem;                     // [128][264] f16 (aliases outf)
  float* outf = (float*)smem;                   // [128][132] f32
  h16*   hh   = (h16*)(smem + 128 * 264 * 2);   // [128][264] f16
  h16*   wbuf = (h16*)(smem + 128 * 264 * 2 * 2);           // [256][40] f16
  float* st   = (float*)(smem + 128 * 264 * 2 * 2 + 256 * 40 * 2);
  const int tid = threadIdx.x, lane = tid & 31, wave = tid >> 5;
  const int nbase = blockIdx.x * 128;
  __builtin_prefetch(w1, 0, 3);

  for (int i = tid; i < 128 * 64; i += 256) {
    int row = i / 64, c4 = (i % 64) * 4;
    int node = min(nbase + row, N_NODES - 1);
    v4h o;
    if (c4 < 128) {
      float4 v = *(const float4*)(V + (size_t)node * CDIM + c4);
      o[0] = (h16)v.x; o[1] = (h16)v.y; o[2] = (h16)v.z; o[3] = (h16)v.w;
    } else {
      int cc = c4 - 128;
      float inv = 1.0f / fmaxf(cnt[node], 1.0f);
      float4 v = *(const float4*)(esum + (size_t)node * CDIM + cc);
      o[0] = (h16)(v.x * inv); o[1] = (h16)(v.y * inv);
      o[2] = (h16)(v.z * inv); o[3] = (h16)(v.w * inv);
    }
    *(v4h*)(xh + (size_t)row * 264 + c4) = o;
  }

  const int mbase = wave * 16, nlane = lane & 15, roff = (lane & 16) ? 8 : 0;
  {                                            // [128x256]@[256->256] + GELU
    v8f acc[16];
#pragma unroll
    for (int nt = 0; nt < 16; ++nt) acc[nt] = (v8f){};
    for (int ks = 0; ks < 8; ++ks) {
      __syncthreads();
      {
        const h16* src = w1 + (size_t)tid * 256 + ks * 32;
        h16* dst = wbuf + tid * 40;
        *(v8h*)(dst +  0) = *(const v8h*)(src +  0);
        *(v8h*)(dst +  8) = *(const v8h*)(src +  8);
        *(v8h*)(dst + 16) = *(const v8h*)(src + 16);
        *(v8h*)(dst + 24) = *(const v8h*)(src + 24);
      }
      __syncthreads();
      v16h a = frag_a(xh, 264, mbase + nlane, ks * 32, lane);
#pragma unroll
      for (int nt = 0; nt < 16; ++nt)
        acc[nt] = wmma16(a, frag_b(wbuf, 40, nt * 16 + nlane, 0, lane), acc[nt]);
    }
#pragma unroll
    for (int nt = 0; nt < 16; ++nt) {
      float bias = b1[nt * 16 + nlane];
      h16* dst = hh + (size_t)(mbase + roff) * 264 + nt * 16 + nlane;
#pragma unroll
      for (int r = 0; r < 8; ++r) dst[(size_t)r * 264] = (h16)gelu_exact(acc[nt][r] + bias);
    }
  }
  {                                            // [128x256]@[256->128]
    v8f acc[8];
#pragma unroll
    for (int nt = 0; nt < 8; ++nt) acc[nt] = (v8f){};
    for (int ks = 0; ks < 8; ++ks) {
      __syncthreads();
      {
        int row = tid >> 1, part = (tid & 1) * 16;
        const h16* src = w2 + (size_t)row * 256 + ks * 32 + part;
        h16* dst = wbuf + row * 40 + part;
        *(v8h*)(dst + 0) = *(const v8h*)(src + 0);
        *(v8h*)(dst + 8) = *(const v8h*)(src + 8);
      }
      __syncthreads();
      v16h a = frag_a(hh, 264, mbase + nlane, ks * 32, lane);
#pragma unroll
      for (int nt = 0; nt < 8; ++nt)
        acc[nt] = wmma16(a, frag_b(wbuf, 40, nt * 16 + nlane, 0, lane), acc[nt]);
    }
#pragma unroll
    for (int nt = 0; nt < 8; ++nt) {
      float bias = b2[nt * 16 + nlane];
      float* dst = outf + (size_t)(mbase + roff) * 132 + nt * 16 + nlane;
#pragma unroll
      for (int r = 0; r < 8; ++r) dst[(size_t)r * 132] = acc[nt][r] + bias;
    }
  }
  __syncthreads();
  for (int r = tid; r < 128; r += 256) {       // LN(v_emb) stats
    const float* row = outf + (size_t)r * 132;
    float s = 0.f, s2 = 0.f;
    for (int c = 0; c < 128; ++c) { float v = row[c]; s += v; s2 += v * v; }
    float mu = s * (1.0f / 128.0f);
    st[r] = mu; st[128 + r] = rsqrtf(s2 * (1.0f / 128.0f) - mu * mu + 1e-5f);
  }
  __syncthreads();
  const float sa1 = silu(a1p[0]);
  for (int i = tid; i < 128 * 128; i += 256) { // Vr = V + silu(a1)*v_emb
    int row = i >> 7, c = i & 127;
    int node = min(nbase + row, N_NODES - 1);
    float vemb = (outf[(size_t)row * 132 + c] - st[row]) * st[128 + row] * g[c] + be[c];
    float vr = V[(size_t)node * CDIM + c] + sa1 * vemb;
    outf[(size_t)row * 132 + c] = vr;
    Vr[(size_t)(nbase + row) * CDIM + c] = vr;
  }
  __syncthreads();
  for (int r = tid; r < 128; r += 256) {       // LN1 stats
    const float* row = outf + (size_t)r * 132;
    float s = 0.f, s2 = 0.f;
    for (int c = 0; c < 128; ++c) { float v = row[c]; s += v; s2 += v * v; }
    float mu = s * (1.0f / 128.0f);
    st[r] = mu; st[128 + r] = rsqrtf(s2 * (1.0f / 128.0f) - mu * mu + 1e-5f);
  }
  __syncthreads();
  for (int i = tid; i < 128 * 128; i += 256) { // W0 = LN1(Vr) -> f16
    int row = i >> 7, c = i & 127;
    float w0 = (outf[(size_t)row * 132 + c] - st[row]) * st[128 + row] * ln1g[c] + ln1b[c];
    W0h[(size_t)(nbase + row) * CDIM + c] = (h16)w0;
  }
}

// ---------------- small scalar kernels ----------------
__global__ void k_q1(const float* __restrict__ Qp, const float* __restrict__ w,
                     const float* __restrict__ b, h16* __restrict__ Q1h) {
  int i = blockIdx.x * blockDim.x + threadIdx.x;
  if (i >= NTOK * CDIM) return;
  int m = i >> 7, c = i & 127;
  float s = b[c];
  for (int k = 0; k < 128; ++k) s += Qp[m * 128 + k] * w[c * 128 + k];
  Q1h[i] = (h16)s;
}

// ---------------- K1/V1 projections via WMMA (k-outer, shared A) ------------
__global__ __launch_bounds__(128)
void k_proj(const h16* __restrict__ W0h,
            const h16* __restrict__ wk, const float* __restrict__ bk,
            const h16* __restrict__ wv, const float* __restrict__ bv,
            h16* __restrict__ K1h, h16* __restrict__ V1Th) {
  const int tid = threadIdx.x, lane = tid & 31, wave = tid >> 5;
  const int nbase = blockIdx.x * 64;
  const int mbase = wave * 16, nlane = lane & 15, roff = (lane & 16) ? 8 : 0;
  const int arow = nbase + mbase + nlane;
  __builtin_prefetch(wk, 0, 3);
  v8f ak[8], av[8];
#pragma unroll
  for (int nt = 0; nt < 8; ++nt) { ak[nt] = (v8f){}; av[nt] = (v8f){}; }
  for (int ks = 0; ks < 4; ++ks) {
    v16h a = frag_a(W0h, CDIM, arow, ks * 32, lane);
#pragma unroll
    for (int nt = 0; nt < 8; ++nt)
      ak[nt] = wmma16(a, frag_b(wk, CDIM, nt * 16 + nlane, ks * 32, lane), ak[nt]);
#pragma unroll
    for (int nt = 0; nt < 8; ++nt)
      av[nt] = wmma16(a, frag_b(wv, CDIM, nt * 16 + nlane, ks * 32, lane), av[nt]);
  }
#pragma unroll
  for (int nt = 0; nt < 8; ++nt) {             // K1 row-major f16
    float bias = bk[nt * 16 + nlane];
#pragma unroll
    for (int r = 0; r < 8; ++r)
      K1h[(size_t)(nbase + mbase + roff + r) * CDIM + nt * 16 + nlane] =
          (h16)(ak[nt][r] + bias);
  }
#pragma unroll
  for (int nt = 0; nt < 8; ++nt) {             // V1 transposed, packed b128 stores
    float bias = bv[nt * 16 + nlane];
    v8h pk;
#pragma unroll
    for (int r = 0; r < 8; ++r) pk[r] = (h16)(av[nt][r] + bias);
    *(v8h*)(V1Th + (size_t)(nt * 16 + nlane) * NPAD + (nbase + mbase + roff)) = pk;
  }
}

// ---------------- attn1: flash-style partials over key chunks ----------------
__global__ __launch_bounds__(128)
void k_attn1(const h16* __restrict__ Q1h, const h16* __restrict__ K1h,
             const h16* __restrict__ V1Th, float* __restrict__ mb_ws,
             float* __restrict__ lb_ws, float* __restrict__ accp) {
  extern __shared__ char smem[];
  float* S = (float*)smem;                     // [64][264] f32
  h16*   P = (h16*)(smem + 64 * 264 * 4);      // [64][264] f16
  const int tid = threadIdx.x, lane = tid & 31, wave = tid >> 5;
  const int kb = blockIdx.x * CHUNK;
  const int mbase = wave * 16, nlane = lane & 15, roff = (lane & 16) ? 8 : 0;
  const float scale = 0.08838834764831845f;    // 128^-0.5
  {                                            // S = Q1 @ K1chunk^T (k-outer)
    v8f acc[16];
#pragma unroll
    for (int nt = 0; nt < 16; ++nt) acc[nt] = (v8f){};
    for (int ks = 0; ks < 4; ++ks) {
      v16h a = frag_a(Q1h, CDIM, mbase + nlane, ks * 32, lane);
#pragma unroll
      for (int nt = 0; nt < 16; ++nt)
        acc[nt] = wmma16(a, frag_b(K1h, CDIM, kb + nt * 16 + nlane, ks * 32, lane),
                         acc[nt]);
    }
#pragma unroll
    for (int nt = 0; nt < 16; ++nt) {
      float* dst = S + (size_t)(mbase + roff) * 264 + nt * 16 + nlane;
#pragma unroll
      for (int r = 0; r < 8; ++r) dst[(size_t)r * 264] = acc[nt][r] * scale;
    }
  }
  __syncthreads();
  const int nvalid = min(CHUNK, N_NODES - kb);
  for (int r = tid; r < 64; r += 128) {
    float* row = S + (size_t)r * 264;
    h16* prow = P + (size_t)r * 264;
    float m = -1e30f;
    for (int c = 0; c < nvalid; ++c) m = fmaxf(m, row[c]);
    float l = 0.f;
    for (int c = 0; c < CHUNK; ++c) {
      float e = (c < nvalid) ? __expf(row[c] - m) : 0.0f;
      l += e;
      prow[c] = (h16)e;
    }
    mb_ws[blockIdx.x * 64 + r] = m;
    lb_ws[blockIdx.x * 64 + r] = l;
  }
  __syncthreads();
  {                                            // acc = P @ V1chunk (k-outer)
    v8f acc[8];
#pragma unroll
    for (int nt = 0; nt < 8; ++nt) acc[nt] = (v8f){};
    for (int ks = 0; ks < 8; ++ks) {
      v16h a = frag_a(P, 264, mbase + nlane, ks * 32, lane);
#pragma unroll
      for (int nt = 0; nt < 8; ++nt)
        acc[nt] = wmma16(a, frag_b(V1Th, NPAD, nt * 16 + nlane, kb + ks * 32, lane),
                         acc[nt]);
    }
#pragma unroll
    for (int nt = 0; nt < 8; ++nt) {
      float* dst = accp + ((size_t)blockIdx.x * 64 + mbase + roff) * CDIM +
                   nt * 16 + nlane;
#pragma unroll
      for (int r = 0; r < 8; ++r) dst[(size_t)r * CDIM] = acc[nt][r];
    }
  }
}

__global__ void k_astats(const float* __restrict__ mb_ws,
                         const float* __restrict__ lb_ws, float* __restrict__ ML) {
  int q = threadIdx.x;
  if (q >= 64) return;
  float M = -1e30f;
  for (int b = 0; b < NCHUNK; ++b) M = fmaxf(M, mb_ws[b * 64 + q]);
  float L = 0.f;
  for (int b = 0; b < NCHUNK; ++b) L += lb_ws[b * 64 + q] * __expf(mb_ws[b * 64 + q] - M);
  ML[q] = M; ML[64 + q] = L;
}

__global__ void k_amerge(const float* __restrict__ accp, const float* __restrict__ mb_ws,
                         const float* __restrict__ ML, float* __restrict__ W1) {
  int i = blockIdx.x * blockDim.x + threadIdx.x;
  if (i >= NTOK * CDIM) return;
  int q = i >> 7;
  float M = ML[q], L = ML[64 + q], s = 0.f;
  for (int b = 0; b < NCHUNK; ++b)
    s += accp[(size_t)b * (NTOK * CDIM) + i] * __expf(mb_ws[b * 64 + q] - M);
  W1[i] = s / L;
}

__global__ void k_qkv(const float* __restrict__ W1, const float* __restrict__ w,
                      const float* __restrict__ b, float* __restrict__ qkv) {
  int i = blockIdx.x * blockDim.x + threadIdx.x;
  if (i >= NTOK * 384) return;
  int t = i / 384, c = i % 384;
  float s = b[c];
  for (int k = 0; k < 128; ++k) s += W1[t * 128 + k] * w[c * 128 + k];
  qkv[i] = s;
}

__global__ void k_mha(const float* __restrict__ qkv, float* __restrict__ o) {
  int head = blockIdx.x, t = threadIdx.x;
  if (t >= 64) return;
  const float sc = 0.17677669529663687f;       // 32^-0.5
  float s[64], m = -1e30f;
  for (int u = 0; u < 64; ++u) {
    float d = 0.f;
    for (int k = 0; k < 32; ++k)
      d += qkv[t * 384 + head * 32 + k] * qkv[u * 384 + 128 + head * 32 + k];
    s[u] = d * sc; m = fmaxf(m, s[u]);
  }
  float l = 0.f;
  for (int u = 0; u < 64; ++u) { s[u] = __expf(s[u] - m); l += s[u]; }
  float inv = 1.0f / l;
  for (int k = 0; k < 32; ++k) {
    float acc = 0.f;
    for (int u = 0; u < 64; ++u) acc += s[u] * qkv[u * 384 + 256 + head * 32 + k];
    o[t * 128 + head * 32 + k] = acc * inv;
  }
}

__global__ void k_w2(const float* __restrict__ o, const float* __restrict__ w,
                     const float* __restrict__ b, float* __restrict__ W2) {
  int i = blockIdx.x * blockDim.x + threadIdx.x;
  if (i >= NTOK * CDIM) return;
  int t = i >> 7, c = i & 127;
  float s = b[c];
  for (int k = 0; k < 128; ++k) s += o[t * 128 + k] * w[c * 128 + k];
  W2[i] = s;
}

__global__ void k_k2v2(const float* __restrict__ W2,
                       const float* __restrict__ kw, const float* __restrict__ kb2,
                       const float* __restrict__ vw, const float* __restrict__ vb2,
                       h16* __restrict__ K2h, h16* __restrict__ V2Th) {
  int i = blockIdx.x * blockDim.x + threadIdx.x;
  if (i >= NTOK * CDIM) return;
  int t = i >> 7, c = i & 127;
  float sk = kb2[c], sv = vb2[c];
  for (int k = 0; k < 128; ++k) {
    float x = W2[t * 128 + k];
    sk += x * kw[c * 128 + k];
    sv += x * vw[c * 128 + k];
  }
  K2h[t * 128 + c] = (h16)sk;
  V2Th[c * 64 + t] = (h16)sv;
}

// ---------------- attn2 + gated residual + LN2 + final linear ----------------
__global__ __launch_bounds__(128)
void k_final(const h16* __restrict__ W0h, const float* __restrict__ Vr,
             const h16* __restrict__ wq2, const float* __restrict__ bq2,
             const h16* __restrict__ K2h, const h16* __restrict__ V2Th,
             const h16* __restrict__ wlin, const float* __restrict__ blin,
             const float* __restrict__ ln2g, const float* __restrict__ ln2b,
             const float* __restrict__ a3p, const float* __restrict__ a4p,
             float* __restrict__ W3) {
  extern __shared__ char smem[];
  h16*   q2h  = (h16*)smem;                              // [64][136] (aliases yh)
  h16*   yh   = q2h;
  float* S2   = (float*)(smem + 64 * 136 * 2);           // [64][72] f32 (aliases zf)
  float* zf   = S2;                                      // [64][132] f32
  h16*   ph   = (h16*)(smem + 64 * 136 * 2 + 33792);     // [64][72] f16
  float* outf = (float*)(smem + 64 * 136 * 2 + 33792 + 64 * 72 * 2); // [64][132]
  float* st   = (float*)(smem + 64 * 136 * 2 + 33792 + 64 * 72 * 2 + 64 * 132 * 4);
  const int tid = threadIdx.x, lane = tid & 31, wave = tid >> 5;
  const int nbase = blockIdx.x * 64;
  const int mbase = wave * 16, nlane = lane & 15, roff = (lane & 16) ? 8 : 0;
  const int arow = nbase + mbase + nlane;
  const float scale = 0.08838834764831845f;
  __builtin_prefetch(wq2, 0, 3);
  __builtin_prefetch(wlin, 0, 3);

  {                                            // Q2 = W0 @ q2_w^T + b (k-outer)
    v8f acc[8];
#pragma unroll
    for (int nt = 0; nt < 8; ++nt) acc[nt] = (v8f){};
    for (int ks = 0; ks < 4; ++ks) {
      v16h a = frag_a(W0h, CDIM, arow, ks * 32, lane);
#pragma unroll
      for (int nt = 0; nt < 8; ++nt)
        acc[nt] = wmma16(a, frag_b(wq2, CDIM, nt * 16 + nlane, ks * 32, lane), acc[nt]);
    }
#pragma unroll
    for (int nt = 0; nt < 8; ++nt) {
      float bias = bq2[nt * 16 + nlane];
      h16* dst = q2h + (size_t)(mbase + roff) * 136 + nt * 16 + nlane;
#pragma unroll
      for (int r = 0; r < 8; ++r) dst[(size_t)r * 136] = (h16)(acc[nt][r] + bias);
    }
  }
  __syncthreads();
  {                                            // S = Q2 @ K2^T (k-outer)
    v8f acc[4];
#pragma unroll
    for (int nt = 0; nt < 4; ++nt) acc[nt] = (v8f){};
    for (int ks = 0; ks < 4; ++ks) {
      v16h a = frag_a(q2h, 136, mbase + nlane, ks * 32, lane);
#pragma unroll
      for (int nt = 0; nt < 4; ++nt)
        acc[nt] = wmma16(a, frag_b(K2h, CDIM, nt * 16 + nlane, ks * 32, lane), acc[nt]);
    }
#pragma unroll
    for (int nt = 0; nt < 4; ++nt) {
      float* dst = S2 + (size_t)(mbase + roff) * 72 + nt * 16 + nlane;
#pragma unroll
      for (int r = 0; r < 8; ++r) dst[(size_t)r * 72] = acc[nt][r] * scale;
    }
  }
  __syncthreads();
  for (int r = tid; r < 64; r += 128) {        // softmax over 64 tokens
    float* row = S2 + (size_t)r * 72;
    h16* prow = ph + (size_t)r * 72;
    float m = -1e30f;
    for (int c = 0; c < 64; ++c) m = fmaxf(m, row[c]);
    float l = 0.f;
    for (int c = 0; c < 64; ++c) { float e = __expf(row[c] - m); l += e; row[c] = e; }
    float inv = 1.0f / l;
    for (int c = 0; c < 64; ++c) prow[c] = (h16)(row[c] * inv);
  }
  __syncthreads();
  {                                            // W3a = P @ V2 (k-outer)
    v8f acc[8];
#pragma unroll
    for (int nt = 0; nt < 8; ++nt) acc[nt] = (v8f){};
    for (int ks = 0; ks < 2; ++ks) {
      v16h a = frag_a(ph, 72, mbase + nlane, ks * 32, lane);
#pragma unroll
      for (int nt = 0; nt < 8; ++nt)
        acc[nt] = wmma16(a, frag_b(V2Th, 64, nt * 16 + nlane, ks * 32, lane), acc[nt]);
    }
#pragma unroll
    for (int nt = 0; nt < 8; ++nt) {
      float* dst = outf + (size_t)(mbase + roff) * 132 + nt * 16 + nlane;
#pragma unroll
      for (int r = 0; r < 8; ++r) dst[(size_t)r * 132] = acc[nt][r];
    }
  }
  __syncthreads();
  const float sa3 = silu(a3p[0]);
  for (int i = tid; i < 64 * 128; i += 128) {  // W2r = Vr + silu(a3)*W3a
    int row = i >> 7, c = i & 127;
    outf[(size_t)row * 132 + c] =
        Vr[(size_t)(nbase + row) * CDIM + c] + sa3 * outf[(size_t)row * 132 + c];
  }
  __syncthreads();
  for (int r = tid; r < 64; r += 128) {        // LN2 stats
    const float* row = outf + (size_t)r * 132;
    float s = 0.f, s2 = 0.f;
    for (int c = 0; c < 128; ++c) { float v = row[c]; s += v; s2 += v * v; }
    float mu = s * (1.0f / 128.0f);
    st[r] = mu; st[64 + r] = rsqrtf(s2 * (1.0f / 128.0f) - mu * mu + 1e-5f);
  }
  __syncthreads();
  for (int i = tid; i < 64 * 128; i += 128) {  // y = LN2(W2r) -> f16
    int row = i >> 7, c = i & 127;
    yh[(size_t)row * 136 + c] =
        (h16)((outf[(size_t)row * 132 + c] - st[row]) * st[64 + row] * ln2g[c] + ln2b[c]);
  }
  __syncthreads();
  {                                            // z = y @ lin_w^T + b (k-outer)
    v8f acc[8];
#pragma unroll
    for (int nt = 0; nt < 8; ++nt) acc[nt] = (v8f){};
    for (int ks = 0; ks < 4; ++ks) {
      v16h a = frag_a(yh, 136, mbase + nlane, ks * 32, lane);
#pragma unroll
      for (int nt = 0; nt < 8; ++nt)
        acc[nt] = wmma16(a, frag_b(wlin, CDIM, nt * 16 + nlane, ks * 32, lane), acc[nt]);
    }
#pragma unroll
    for (int nt = 0; nt < 8; ++nt) {
      float bias = blin[nt * 16 + nlane];
      float* dst = zf + (size_t)(mbase + roff) * 132 + nt * 16 + nlane;
#pragma unroll
      for (int r = 0; r < 8; ++r) dst[(size_t)r * 132] = acc[nt][r] + bias;
    }
  }
  __syncthreads();
  const float sa4 = silu(a4p[0]);
  for (int i = tid; i < 64 * 32; i += 128) {   // W3 = W2r + silu(a4)*z
    int row = i / 32, c4 = (i % 32) * 4;
    int grow = nbase + row;
    if (grow < N_NODES) {
      float4 o;
      o.x = outf[(size_t)row * 132 + c4 + 0] + sa4 * zf[(size_t)row * 132 + c4 + 0];
      o.y = outf[(size_t)row * 132 + c4 + 1] + sa4 * zf[(size_t)row * 132 + c4 + 1];
      o.z = outf[(size_t)row * 132 + c4 + 2] + sa4 * zf[(size_t)row * 132 + c4 + 2];
      o.w = outf[(size_t)row * 132 + c4 + 3] + sa4 * zf[(size_t)row * 132 + c4 + 3];
      *(float4*)(W3 + (size_t)grow * CDIM + c4) = o;
    }
  }
}

// ---------------- workspace layout (bytes, 256-aligned) ----------------
static constexpr size_t OFF_WFE1 = 0;                          // 256x384 f16
static constexpr size_t OFF_WFE2 = OFF_WFE1 + 196608;          // 128x256 f16
static constexpr size_t OFF_WFN1 = OFF_WFE2 + 65536;           // 256x256 f16
static constexpr size_t OFF_WFN2 = OFF_WFN1 + 131072;          // 128x256 f16
static constexpr size_t OFF_WK1  = OFF_WFN2 + 65536;
static constexpr size_t OFF_WV1  = OFF_WK1 + 32768;
static constexpr size_t OFF_WQ2  = OFF_WV1 + 32768;
static constexpr size_t OFF_WLIN = OFF_WQ2 + 32768;
static constexpr size_t OFF_ESUM = OFF_WLIN + 32768;           // NPAD*128 f32
static constexpr size_t OFF_CNT  = OFF_ESUM + (size_t)NPAD * 128 * 4; // NPAD f32
static constexpr size_t OFF_VR   = OFF_CNT + (size_t)NPAD * 4;
static constexpr size_t OFF_W0H  = OFF_VR + (size_t)NPAD * 128 * 4;
static constexpr size_t OFF_K1H  = OFF_W0H + (size_t)NPAD * 128 * 2;
static constexpr size_t OFF_V1T  = OFF_K1H + (size_t)NPAD * 128 * 2;
static constexpr size_t OFF_Q1H  = OFF_V1T + (size_t)NPAD * 128 * 2;
static constexpr size_t OFF_MB   = OFF_Q1H + 16384;
static constexpr size_t OFF_LB   = OFF_MB + (size_t)NCHUNK * 64 * 4;
static constexpr size_t OFF_ACCP = OFF_LB + (size_t)NCHUNK * 64 * 4;
static constexpr size_t OFF_ML   = OFF_ACCP + (size_t)NCHUNK * 64 * 128 * 4;
static constexpr size_t OFF_QKV  = OFF_ML + 512;
static constexpr size_t OFF_MO   = OFF_QKV + 98304;
static constexpr size_t OFF_W1   = OFF_MO + 32768;
static constexpr size_t OFF_W2   = OFF_W1 + 32768;
static constexpr size_t OFF_K2H  = OFF_W2 + 32768;
static constexpr size_t OFF_V2T  = OFF_K2H + 16384;

extern "C" void kernel_launch(void* const* d_in, const int* in_sizes, int n_in,
                              void* d_out, int out_size, void* d_ws, size_t ws_size,
                              hipStream_t stream) {
  (void)in_sizes; (void)n_in; (void)out_size; (void)ws_size;
  const float* V        = (const float*)d_in[0];
  const float* E        = (const float*)d_in[1];
  const int*   edges    = (const int*)d_in[2];
  const float* fe_w1 = (const float*)d_in[4];  const float* fe_b1 = (const float*)d_in[5];
  const float* fe_w2 = (const float*)d_in[6];  const float* fe_b2 = (const float*)d_in[7];
  const float* fe_g  = (const float*)d_in[8];  const float* fe_be = (const float*)d_in[9];
  const float* fn_w1 = (const float*)d_in[10]; const float* fn_b1 = (const float*)d_in[11];
  const float* fn_w2 = (const float*)d_in[12]; const float* fn_b2 = (const float*)d_in[13];
  const float* fn_g  = (const float*)d_in[14]; const float* fn_be = (const float*)d_in[15];
  const float* ln1_g = (const float*)d_in[16]; const float* ln1_b = (const float*)d_in[17];
  const float* ln2_g = (const float*)d_in[18]; const float* ln2_b = (const float*)d_in[19];
  const float* lin_w = (const float*)d_in[20]; const float* lin_b = (const float*)d_in[21];
  const float* Qp    = (const float*)d_in[22];
  const float* q1_w  = (const float*)d_in[23]; const float* q1_b = (const float*)d_in[24];
  const float* k1_w  = (const float*)d_in[25]; const float* k1_b = (const float*)d_in[26];
  const float* v1_w  = (const float*)d_in[27]; const float* v1_b = (const float*)d_in[28];
  const float* mha_in_w  = (const float*)d_in[29]; const float* mha_in_b  = (const float*)d_in[30];
  const float* mha_out_w = (const float*)d_in[31]; const float* mha_out_b = (const float*)d_in[32];
  const float* q2_w  = (const float*)d_in[33]; const float* q2_b = (const float*)d_in[34];
  const float* k2_w  = (const float*)d_in[35]; const float* k2_b = (const float*)d_in[36];
  const float* v2_w  = (const float*)d_in[37]; const float* v2_b = (const float*)d_in[38];
  const float* a1 = (const float*)d_in[39]; const float* a2 = (const float*)d_in[40];
  const float* a3 = (const float*)d_in[41]; const float* a4 = (const float*)d_in[42];

  float* W3   = (float*)d_out;
  float* Eout = (float*)d_out + (size_t)N_NODES * CDIM;
  char* ws = (char*)d_ws;
  h16* wh_fe1 = (h16*)(ws + OFF_WFE1); h16* wh_fe2 = (h16*)(ws + OFF_WFE2);
  h16* wh_fn1 = (h16*)(ws + OFF_WFN1); h16* wh_fn2 = (h16*)(ws + OFF_WFN2);
  h16* wh_k1  = (h16*)(ws + OFF_WK1);  h16* wh_v1  = (h16*)(ws + OFF_WV1);
  h16* wh_q2  = (h16*)(ws + OFF_WQ2);  h16* wh_lin = (h16*)(ws + OFF_WLIN);
  float* esum = (float*)(ws + OFF_ESUM); float* cnt = (float*)(ws + OFF_CNT);
  float* Vr   = (float*)(ws + OFF_VR);
  h16* W0h = (h16*)(ws + OFF_W0H); h16* K1h = (h16*)(ws + OFF_K1H);
  h16* V1Th = (h16*)(ws + OFF_V1T); h16* Q1h = (h16*)(ws + OFF_Q1H);
  float* mb_ws = (float*)(ws + OFF_MB); float* lb_ws = (float*)(ws + OFF_LB);
  float* accp = (float*)(ws + OFF_ACCP); float* ML = (float*)(ws + OFF_ML);
  float* qkv = (float*)(ws + OFF_QKV); float* mo = (float*)(ws + OFF_MO);
  float* W1f = (float*)(ws + OFF_W1); float* W2f = (float*)(ws + OFF_W2);
  h16* K2h = (h16*)(ws + OFF_K2H); h16* V2Th = (h16*)(ws + OFF_V2T);

  const size_t SMEM_EDGE  = 128 * 392 * 2 + 128 * 264 * 2 + 256 * 40 * 2 + 1024; // 189440
  const size_t SMEM_NODE  = 128 * 264 * 2 * 2 + 256 * 40 * 2 + 1024;             // 156672
  const size_t SMEM_ATTN  = 64 * 264 * 4 + 64 * 264 * 2;                         // 101376
  const size_t SMEM_FINAL = 64 * 136 * 2 + 33792 + 64 * 72 * 2 + 33792 + 512;    // 94720
  (void)hipFuncSetAttribute((const void*)k_edge,  hipFuncAttributeMaxDynamicSharedMemorySize, (int)SMEM_EDGE);
  (void)hipFuncSetAttribute((const void*)k_node,  hipFuncAttributeMaxDynamicSharedMemorySize, (int)SMEM_NODE);
  (void)hipFuncSetAttribute((const void*)k_attn1, hipFuncAttributeMaxDynamicSharedMemorySize, (int)SMEM_ATTN);
  (void)hipFuncSetAttribute((const void*)k_final, hipFuncAttributeMaxDynamicSharedMemorySize, (int)SMEM_FINAL);

  // weight conversion f32 -> f16
  k_cvt<<<(98304 + 255) / 256, 256, 0, stream>>>(fe_w1, wh_fe1, 98304);
  k_cvt<<<(32768 + 255) / 256, 256, 0, stream>>>(fe_w2, wh_fe2, 32768);
  k_cvt<<<(65536 + 255) / 256, 256, 0, stream>>>(fn_w1, wh_fn1, 65536);
  k_cvt<<<(32768 + 255) / 256, 256, 0, stream>>>(fn_w2, wh_fn2, 32768);
  k_cvt<<<(16384 + 255) / 256, 256, 0, stream>>>(k1_w, wh_k1, 16384);
  k_cvt<<<(16384 + 255) / 256, 256, 0, stream>>>(v1_w, wh_v1, 16384);
  k_cvt<<<(16384 + 255) / 256, 256, 0, stream>>>(q2_w, wh_q2, 16384);
  k_cvt<<<(16384 + 255) / 256, 256, 0, stream>>>(lin_w, wh_lin, 16384);
  // zero scatter accumulators (esum + cnt contiguous)
  {
    long nz = (long)NPAD * 129;
    k_zero<<<(unsigned)((nz + 255) / 256), 256, 0, stream>>>(esum, nz);
  }
  k_q1<<<32, 256, 0, stream>>>(Qp, q1_w, q1_b, Q1h);
  k_edge<<<GRID_E, 256, SMEM_EDGE, stream>>>(V, E, edges, wh_fe1, fe_b1,
      wh_fe2, fe_b2, fe_g, fe_be, a2, Eout, esum, cnt);
  k_node<<<GRID_ND, 256, SMEM_NODE, stream>>>(V, esum, cnt, wh_fn1, fn_b1, wh_fn2,
      fn_b2, fn_g, fn_be, ln1_g, ln1_b, a1, Vr, W0h);
  k_proj<<<GRID_N, 128, 0, stream>>>(W0h, wh_k1, k1_b, wh_v1, v1_b, K1h, V1Th);
  k_attn1<<<NCHUNK, 128, SMEM_ATTN, stream>>>(Q1h, K1h, V1Th, mb_ws, lb_ws, accp);
  k_astats<<<1, 64, 0, stream>>>(mb_ws, lb_ws, ML);
  k_amerge<<<32, 256, 0, stream>>>(accp, mb_ws, ML, W1f);
  k_qkv<<<96, 256, 0, stream>>>(W1f, mha_in_w, mha_in_b, qkv);
  k_mha<<<4, 64, 0, stream>>>(qkv, mo);
  k_w2<<<32, 256, 0, stream>>>(mo, mha_out_w, mha_out_b, W2f);
  k_k2v2<<<32, 256, 0, stream>>>(W2f, k2_w, k2_b, v2_w, v2_b, K2h, V2Th);
  k_final<<<GRID_N, 128, SMEM_FINAL, stream>>>(W0h, Vr, wh_q2, q2_b, K2h, V2Th,
      wh_lin, lin_b, ln2_g, ln2_b, a3, a4, W3);
}
